// Head_64149631533203
// MI455X (gfx1250) — matrix operand order repaired and checked by reading, not compiled, not run on hardware
//
#include <hip/hip_runtime.h>

typedef __attribute__((ext_vector_type(16))) _Float16 v16h;
typedef __attribute__((ext_vector_type(8)))  _Float16 v8h;
typedef __attribute__((ext_vector_type(8)))  float    v8f;

#define Bsz 1024
#define Tsz 256
#define Csz 512
#define Hsz 64

__device__ __forceinline__ v8f wmma16(v16h a, v16h b, v8f c) {
  // D = A(16x32 f16) x B(32x16 f16) + C(16x16 f32)
  return __builtin_amdgcn_wmma_f32_16x16x32_f16(
      /*neg_a=*/false, a, /*neg_b=*/false, b,
      /*c_mod=*/(short)0, c, /*reuse_a=*/false, /*reuse_b=*/false);
}

// ---------------------------------------------------------------------------
// Kernel 0: pre-swizzle Wq/Wk/Wv (fp32 [C,H]) into per-lane WMMA B-fragments.
// bpack layout: ((mat*4 + ntile)*16 + kstep) * 512 + lane*16 + j   (f16)
// B-frag: lane -> column N = ntile*16 + (lane&15); half j -> K = kstep*32 + j + 16*(lane>=16)
// ---------------------------------------------------------------------------
__global__ void pack_w_kernel(const float* __restrict__ Wq, const float* __restrict__ Wk,
                              const float* __restrict__ Wv, _Float16* __restrict__ bpack) {
  const int id   = blockIdx.x;            // 0..191
  const int lane = threadIdx.x;           // 32 threads
  const int mat  = id >> 6;
  const float* W = (mat == 0) ? Wq : ((mat == 1) ? Wk : Wv);
  const int nt   = (id >> 4) & 3;
  const int ks   = id & 15;
  const int khi  = (lane >= 16) ? 16 : 0;
  const int n    = nt * 16 + (lane & 15);
  _Float16* dst  = bpack + (size_t)id * 512 + lane * 16;
#pragma unroll
  for (int j = 0; j < 16; ++j) {
    const int kk = ks * 32 + khi + j;
    dst[j] = (_Float16)W[kk * Hsz + n];
  }
}

// ---------------------------------------------------------------------------
// Kernel 1: Q/K/V projections, M=32 per wave (two 16-row A tiles share every
// B fragment -> 2 back-to-back WMMAs per B load, half the B traffic).
// Needs ~240 VGPRs: raise the register budget (waves_per_eu min=2) so the
// 24 f32 accumulator tiles stay in registers instead of spilling to scratch.
// Qh/Kh: f16 [B*T, 64] row-major (Q pre-scaled by 1/sqrt(64)).
// Vt:    f16 [B, 64, T] (transposed within each batch) for the PV A-fragments.
// ---------------------------------------------------------------------------
__global__ void __launch_bounds__(256)
__attribute__((amdgpu_waves_per_eu(2, 8)))
proj_qkv_kernel(const float* __restrict__ x, const _Float16* __restrict__ bpack,
                _Float16* __restrict__ Qh, _Float16* __restrict__ Kh,
                _Float16* __restrict__ Vt) {
  const int lane   = threadIdx.x & 31;
  const int wave   = threadIdx.x >> 5;
  const int tile32 = blockIdx.x * 8 + wave;      // 0 .. B*T/32 - 1
  const int l15    = lane & 15;
  const int hi8    = (lane >= 16) ? 8 : 0;
  const int t0     = tile32 * 32;                // first token of this 32-row tile
  const float* xr0 = x + ((size_t)t0 + l15) * Csz;
  const float* xr1 = xr0 + (size_t)16 * Csz;

  v8f cq[2][4] = {}, ck[2][4] = {}, cv[2][4] = {};

  for (int ks = 0; ks < 16; ++ks) {              // K loop over C=512, step 32
    const int kb = ks * 32;
    // Two A frags (16x32 f16): halves K = kb+2*hi8+[0,8) and kb+2*hi8+8+[0,8)
    v16h a0, a1;
    {
      const float* p00 = xr0 + kb + 2 * hi8;
      const float* p01 = xr0 + kb + 8 + 2 * hi8;
      const float* p10 = xr1 + kb + 2 * hi8;
      const float* p11 = xr1 + kb + 8 + 2 * hi8;
#pragma unroll
      for (int j = 0; j < 8; ++j) {
        a0[j] = (_Float16)p00[j]; a0[j + 8] = (_Float16)p01[j];
        a1[j] = (_Float16)p10[j]; a1[j + 8] = (_Float16)p11[j];
      }
    }
    const _Float16* bp = bpack + (size_t)ks * 512 + lane * 16;
#pragma unroll
    for (int nt = 0; nt < 4; ++nt) {
      const v16h bq = *(const v16h*)(bp + ((size_t)(0 * 4 + nt) * 16) * 512);
      cq[0][nt] = wmma16(a0, bq, cq[0][nt]);
      cq[1][nt] = wmma16(a1, bq, cq[1][nt]);
      const v16h bk = *(const v16h*)(bp + ((size_t)(1 * 4 + nt) * 16) * 512);
      ck[0][nt] = wmma16(a0, bk, ck[0][nt]);
      ck[1][nt] = wmma16(a1, bk, ck[1][nt]);
      const v16h bv = *(const v16h*)(bp + ((size_t)(2 * 4 + nt) * 16) * 512);
      cv[0][nt] = wmma16(a0, bv, cv[0][nt]);
      cv[1][nt] = wmma16(a1, bv, cv[1][nt]);
    }
  }

  // C-frag: lane -> column h = nt*16+l15; VGPR r -> row t = t0 + i*16 + hi8 + r
  const int b    = t0 >> 8;                      // batch (T=256)
  const int tl0  = t0 & 255;                     // token offset within batch
#pragma unroll
  for (int i = 0; i < 2; ++i) {
    const int tbase = t0 + i * 16 + hi8;
#pragma unroll
    for (int nt = 0; nt < 4; ++nt) {
      const int h = nt * 16 + l15;
#pragma unroll
      for (int r = 0; r < 8; ++r) {
        const size_t tt = (size_t)(tbase + r);
        Qh[tt * Hsz + h] = (_Float16)(cq[i][nt][r] * 0.125f);  // fold 1/sqrt(64)
        Kh[tt * Hsz + h] = (_Float16)ck[i][nt][r];
      }
    }
    // V transposed per batch: Vt[b][h][t], contiguous in t -> one b128 store
    const int tloc = tl0 + i * 16 + hi8;
#pragma unroll
    for (int nt = 0; nt < 4; ++nt) {
      const int h = nt * 16 + l15;
      v8h pk;
#pragma unroll
      for (int r = 0; r < 8; ++r) pk[r] = (_Float16)cv[i][nt][r];
      *(v8h*)(Vt + ((size_t)b * Hsz + h) * Tsz + tloc) = pk;
    }
  }
}

// ---------------------------------------------------------------------------
// Kernel 2: causal flash attention. One wave per 16-query tile.
// S^T = K * Q^T (softmax reduction in-lane + one half-wave shfl), then
// out^T = V^T * P^T with online-softmax rescaling. All K and V fragment
// loads for a 32-key step are issued up front so they overlap the exp VALU.
// ---------------------------------------------------------------------------
__global__ void __launch_bounds__(256)
__attribute__((amdgpu_waves_per_eu(2, 8)))
attn_kernel(const _Float16* __restrict__ Qh, const _Float16* __restrict__ Kh,
            const _Float16* __restrict__ Vt, float* __restrict__ out) {
  const int lane = threadIdx.x & 31;
  const int wave = threadIdx.x >> 5;
  const int tile = blockIdx.x * 8 + wave;        // 0 .. B*(T/16) - 1
  const int l15  = lane & 15;
  const bool hiw = lane >= 16;
  const int hi8  = hiw ? 8 : 0;
  const int b    = tile >> 4;
  const int qt   = tile & 15;
  const int qloc = qt * 16 + l15;                // this lane's query row
  const size_t qflat = (size_t)b * Tsz + qloc;

  // Q^T B-fragments (loop-invariant): lane -> query column; halves -> h dim
  const v16h qb0 = *(const v16h*)(Qh + qflat * Hsz + (hiw ? 16 : 0));
  const v16h qb1 = *(const v16h*)(Qh + qflat * Hsz + 32 + (hiw ? 16 : 0));

  const _Float16* Kb = Kh + (size_t)b * Tsz * Hsz;
  const _Float16* Vb = Vt + (size_t)b * Hsz * Tsz;

  v8f acc[4] = {};                               // out^T accumulators (4 h-tiles)
  float m = -1e30f, l = 0.f;
  const int nsteps = qt / 2 + 1;                 // causal: keys 0 .. qt*16+15

  for (int s2 = 0; s2 < nsteps; ++s2) {
    const int key0 = s2 * 32;

    // ---- issue ALL loads for this 32-key step up front -------------------
    // K A-frags: rows = keys key0+l15 (tile0) / key0+16+l15 (tile1),
    //            halves = h at kb*32 + 2*hi8 + [0,8) and +8+[0,8)
    const _Float16* kr0 = Kb + (size_t)(key0 + l15) * Hsz + 2 * hi8;
    const _Float16* kr1 = Kb + (size_t)(key0 + 16 + l15) * Hsz + 2 * hi8;
    v8h klo0[2], khi0[2], klo1[2], khi1[2];
#pragma unroll
    for (int kb = 0; kb < 2; ++kb) {
      klo0[kb] = *(const v8h*)(kr0 + kb * 32);
      khi0[kb] = *(const v8h*)(kr0 + kb * 32 + 8);
      klo1[kb] = *(const v8h*)(kr1 + kb * 32);
      khi1[kb] = *(const v8h*)(kr1 + kb * 32 + 8);
    }
    // V^T A-frags (independent of softmax -> overlaps the exp work)
    v8h vlo[4], vhi[4];
#pragma unroll
    for (int ht = 0; ht < 4; ++ht) {
      const _Float16* vr = Vb + (size_t)(ht * 16 + l15) * Tsz + key0 + 2 * hi8;
      vlo[ht] = *(const v8h*)vr;
      vhi[ht] = *(const v8h*)(vr + 8);
    }

    // ---- S^T = K * Q^T ----------------------------------------------------
    v8f c0 = {}, c1 = {};                        // keys [key0,+16), [key0+16,+32)
#pragma unroll
    for (int kb = 0; kb < 2; ++kb) {
      v16h a0, a1;
#pragma unroll
      for (int j = 0; j < 8; ++j) {
        a0[j] = klo0[kb][j]; a0[j + 8] = khi0[kb][j];
        a1[j] = klo1[kb][j]; a1[j + 8] = khi1[kb][j];
      }
      const v16h qb = kb ? qb1 : qb0;
      c0 = wmma16(a0, qb, c0);
      c1 = wmma16(a1, qb, c1);
    }

    // ---- causal mask + online softmax (S^T: VGPR r -> key, lane -> query) -
    float tm = -1e30f;
#pragma unroll
    for (int r = 0; r < 8; ++r) {
      const int k0 = key0 + hi8 + r;
      const int k1 = key0 + 16 + hi8 + r;
      const float s0 = (k0 <= qloc) ? c0[r] : -1e30f;
      const float s1 = (k1 <= qloc) ? c1[r] : -1e30f;
      c0[r] = s0; c1[r] = s1;
      tm = fmaxf(tm, fmaxf(s0, s1));
    }
    tm = fmaxf(tm, __shfl_xor(tm, 16, 32));      // combine lane-halves (same query)
    const float mnew  = fmaxf(m, tm);
    const float alpha = __expf(m - mnew);
    float e0[8], e1[8];
    float rs = 0.f;
#pragma unroll
    for (int r = 0; r < 8; ++r) {
      e0[r] = __expf(c0[r] - mnew);
      e1[r] = __expf(c1[r] - mnew);
      rs += e0[r] + e1[r];
    }
    rs += __shfl_xor(rs, 16, 32);
    l = l * alpha + rs;
    m = mnew;
#pragma unroll
    for (int nt = 0; nt < 4; ++nt)
#pragma unroll
      for (int r = 0; r < 8; ++r) acc[nt][r] *= alpha;

    // P^T B-frag: lane -> query; halves j -> key = key0 + j + 16*(lane>=16)
    v16h pb;
#pragma unroll
    for (int r = 0; r < 8; ++r) {
      const float o0 = __shfl_xor(e0[r], 16, 32);
      const float o1 = __shfl_xor(e1[r], 16, 32);
      pb[r]     = (_Float16)(hiw ? o1 : e0[r]);  // keys key0+[0,8) / key0+16+[0,8)
      pb[r + 8] = (_Float16)(hiw ? e1[r] : o0);  // keys key0+[8,16) / key0+24+[0,8)
    }

    // ---- out^T += V^T * P^T (V frags already resident) --------------------
#pragma unroll
    for (int ht = 0; ht < 4; ++ht) {
      v16h av;
#pragma unroll
      for (int j = 0; j < 8; ++j) { av[j] = vlo[ht][j]; av[j + 8] = vhi[ht][j]; }
      acc[ht] = wmma16(av, pb, acc[ht]);
    }
  }

  // out^T D-frag: lane -> query, VGPR r -> h = ht*16 + hi8 + r  (contiguous store)
  const float inv = 1.0f / l;
  float* orow = out + qflat * Hsz;
#pragma unroll
  for (int ht = 0; ht < 4; ++ht) {
    v8f o;
#pragma unroll
    for (int r = 0; r < 8; ++r) o[r] = acc[ht][r] * inv;
    *(v8f*)(orow + ht * 16 + hi8) = o;
  }
}

// ---------------------------------------------------------------------------
extern "C" void kernel_launch(void* const* d_in, const int* in_sizes, int n_in,
                              void* d_out, int out_size, void* d_ws, size_t ws_size,
                              hipStream_t stream) {
  const float* x  = (const float*)d_in[0];
  const float* Wq = (const float*)d_in[1];
  const float* Wk = (const float*)d_in[2];
  const float* Wv = (const float*)d_in[3];
  float* out = (float*)d_out;

  // Workspace: Qh | Kh | Vt (each B*T*H f16 = 32 MiB) | bpack (192 KiB)
  const size_t qkvElems = (size_t)Bsz * Tsz * Hsz;  // 16,777,216
  _Float16* Qh    = (_Float16*)d_ws;
  _Float16* Kh    = Qh + qkvElems;
  _Float16* Vt    = Kh + qkvElems;
  _Float16* bpack = Vt + qkvElems;

  hipLaunchKernelGGL(pack_w_kernel, dim3(3 * 4 * 16), dim3(32), 0, stream, Wq, Wk, Wv, bpack);

  const int tiles32 = Bsz * Tsz / 32;               // 8192 tiles, 8 waves/block
  hipLaunchKernelGGL(proj_qkv_kernel, dim3(tiles32 / 8), dim3(256), 0, stream, x, bpack, Qh, Kh, Vt);

  const int qtiles = Bsz * (Tsz / 16);              // 16384 q-tiles
  hipLaunchKernelGGL(attn_kernel, dim3(qtiles / 8), dim3(256), 0, stream, Qh, Kh, Vt, out);
}
